// HybridGNN_8787503088150
// MI455X (gfx1250) — matrix operand order repaired and checked by reading, not compiled, not run on hardware
//
#include <hip/hip_runtime.h>
#include <hip/hip_bf16.h>
#include <cfloat>

typedef __attribute__((ext_vector_type(16))) _Float16 v16h;
typedef __attribute__((ext_vector_type(8)))  _Float16 v8h;
typedef __attribute__((ext_vector_type(8)))  float    v8f;

#define HID 128
#define EPS 1e-5f
#define LDK 136  // padded K-stride in halves: 272B rows -> 16B aligned, low-conflict

#define SHUF16(lo, hi) \
  __builtin_shufflevector(lo, hi, 0, 1, 2, 3, 4, 5, 6, 7, 8, 9, 10, 11, 12, 13, 14, 15)

__device__ inline v8f wmma16(v16h a, v16h b, v8f c) {
  return __builtin_amdgcn_wmma_f32_16x16x32_f16(false, a, false, b, (short)0, c,
                                                false, false);
}

// ---------------------------------------------------------------------------
// GEMM: Out[nrows x 128] = A[nrows x K, ld=lda] * W[K x 128]  (f32 in/out)
// Split-precision f16 WMMA (a = a_hi + a_lo): Ah*Bh + Ah*Bl + Al*Bh, f32 acc.
// Both A-tile (16 rows) and W-chunk are converted once per 128-k chunk into
// LDS, W transposed to [n][k] so every fragment is contiguous ds_load_b128.
// ---------------------------------------------------------------------------
__global__ void __launch_bounds__(256)
gemm_wmma_kernel(const float* __restrict__ A, int lda, int K,
                 const float* __restrict__ W, float* __restrict__ Out, int nrows) {
  __shared__ __align__(16) _Float16 Bh[128 * LDK];  // W hi, [n][k]
  __shared__ __align__(16) _Float16 Bl[128 * LDK];  // W lo
  __shared__ __align__(16) _Float16 Ah[16 * LDK];   // A-tile hi, [m][k]
  __shared__ __align__(16) _Float16 Al[16 * LDK];   // A-tile lo
  const int wave   = threadIdx.x >> 5;
  const int lane   = threadIdx.x & 31;
  const int laneM  = lane & 15;
  const int halfId = lane >> 4;
  const int m0 = blockIdx.x * 16;
  const int n0 = wave * 16;

  v8f c = {};
  for (int kc = 0; kc < K; kc += 128) {
    __syncthreads();
    // stage W chunk, transposed + split (coalesced global reads over n)
    for (int idx = threadIdx.x; idx < 128 * 128; idx += 256) {
      int k = idx >> 7, n = idx & 127;
      float w = W[(size_t)(kc + k) * 128 + n];
      _Float16 hi = (_Float16)w;
      Bh[n * LDK + k] = hi;
      Bl[n * LDK + k] = (_Float16)(w - (float)hi);
    }
    // stage the block's 16 A rows, split (coalesced global reads over k)
    for (int idx = threadIdx.x; idx < 16 * 128; idx += 256) {
      int m = idx >> 7, k = idx & 127;
      int r = m0 + m;
      if (r >= nrows) r = nrows - 1;         // clamp loads; stores are guarded
      float v = A[(size_t)r * lda + kc + k];
      _Float16 hi = (_Float16)v;
      Ah[m * LDK + k] = hi;
      Al[m * LDK + k] = (_Float16)(v - (float)hi);
    }
    __syncthreads();
#pragma unroll
    for (int k0 = 0; k0 < 128; k0 += 32) {
      // 16-bit A 16x32 layout: lane-half h -> K {0..7,16..23} + 8h (contiguous runs)
      const int aoff = laneM * LDK + k0 + 8 * halfId;
      v8h a0h = *(const v8h*)(Ah + aoff);
      v8h a1h = *(const v8h*)(Ah + aoff + 16);
      v8h a0l = *(const v8h*)(Al + aoff);
      v8h a1l = *(const v8h*)(Al + aoff + 16);
      // 16-bit B 32x16 layout: lane-half h -> K {0..7,16..23} + 8h (same as A)
      const int boff = (n0 + laneM) * LDK + k0 + 8 * halfId;
      v8h b0h = *(const v8h*)(Bh + boff);
      v8h b1h = *(const v8h*)(Bh + boff + 16);
      v8h b0l = *(const v8h*)(Bl + boff);
      v8h b1l = *(const v8h*)(Bl + boff + 16);
      v16h ah = SHUF16(a0h, a1h), al = SHUF16(a0l, a1l);
      v16h bh = SHUF16(b0h, b1h), bl = SHUF16(b0l, b1l);
      c = wmma16(ah, bh, c);
      c = wmma16(ah, bl, c);
      c = wmma16(al, bh, c);
    }
  }
#pragma unroll
  for (int r = 0; r < 8; ++r) {              // C/D: VGPR r -> M = r + 8*half
    int orow = m0 + halfId * 8 + r;
    if (orow < nrows) Out[(size_t)orow * 128 + n0 + laneM] = c[r];
  }
}

// ---------------------------------------------------------------------------
// Utility / sparse kernels
// ---------------------------------------------------------------------------
__global__ void fill_kernel(float* __restrict__ p, float v, long long n) {
  long long i = (long long)blockIdx.x * blockDim.x + threadIdx.x;
  if (i < n) p[i] = v;
}

__global__ void deg_kernel(const int* __restrict__ ei, int E, int N,
                           float* __restrict__ deg) {
  int i = blockIdx.x * blockDim.x + threadIdx.x;
  if (i >= E + N) return;
  int d = i < E ? ei[E + i] : i - E;         // self loops appended
  atomicAdd(&deg[d], 1.0f);
}

__global__ void dinv_kernel(const float* __restrict__ deg,
                            float* __restrict__ dinv, int N) {
  int i = blockIdx.x * blockDim.x + threadIdx.x;
  if (i >= N) return;
  float d = deg[i];
  dinv[i] = d > 0.f ? rsqrtf(d) : 0.f;
}

// one wave per edge, lane handles 4 channels (float4 gather, 4 float atomics)
__global__ void gcn_scatter_kernel(const int* __restrict__ ei, int E, int N,
                                   const float* __restrict__ tmp,
                                   const float* __restrict__ dinv,
                                   float* __restrict__ agg) {
  long long idx = (long long)blockIdx.x * blockDim.x + threadIdx.x;
  int edge = (int)(idx >> 5);
  int lane = (int)(idx & 31);
  if (edge >= E + N) return;
  int s = edge < E ? ei[edge]     : edge - E;
  int d = edge < E ? ei[E + edge] : edge - E;
  float norm = dinv[s] * dinv[d];
  float4 v = ((const float4*)(tmp + (size_t)s * 128))[lane];
  float* a = agg + (size_t)d * 128 + lane * 4;
  atomicAdd(a + 0, v.x * norm);
  atomicAdd(a + 1, v.y * norm);
  atomicAdd(a + 2, v.z * norm);
  atomicAdd(a + 3, v.w * norm);
}

__global__ void gat_att_kernel(const float* __restrict__ tmp,
                               const float* __restrict__ a_src,
                               const float* __restrict__ a_dst,
                               float* __restrict__ asn, float* __restrict__ adn,
                               int N) {
  int i = blockIdx.x * blockDim.x + threadIdx.x;  // (node, head)
  if (i >= N * 4) return;
  int h = i & 3;
  const float* row = tmp + (size_t)(i >> 2) * 128 + h * 32;
  const float* ws = a_src + h * 32;
  const float* wd = a_dst + h * 32;
  float s = 0.f, d = 0.f;
  for (int c = 0; c < 32; ++c) { s += row[c] * ws[c]; d += row[c] * wd[c]; }
  asn[i] = s;
  adn[i] = d;
}

__device__ inline void atomicMaxFloat(float* addr, float val) {
  if (val >= 0.f) atomicMax((int*)addr, __float_as_int(val));
  else            atomicMin((unsigned int*)addr, __float_as_uint(val));
}

__global__ void gat_max_kernel(const int* __restrict__ ei, int E, int N,
                               const float* __restrict__ asn,
                               const float* __restrict__ adn,
                               float* __restrict__ mseg) {
  long long idx = (long long)blockIdx.x * blockDim.x + threadIdx.x;
  int edge = (int)(idx >> 2), h = (int)(idx & 3);
  if (edge >= E + N) return;
  int s = edge < E ? ei[edge]     : edge - E;
  int d = edge < E ? ei[E + edge] : edge - E;
  float e = asn[s * 4 + h] + adn[d * 4 + h];
  e = e > 0.f ? e : 0.2f * e;                 // leaky_relu(0.2)
  atomicMaxFloat(&mseg[d * 4 + h], e);
}

__global__ void gat_sum_kernel(const int* __restrict__ ei, int E, int N,
                               const float* __restrict__ asn,
                               const float* __restrict__ adn,
                               const float* __restrict__ mseg,
                               float* __restrict__ sseg) {
  long long idx = (long long)blockIdx.x * blockDim.x + threadIdx.x;
  int edge = (int)(idx >> 2), h = (int)(idx & 3);
  if (edge >= E + N) return;
  int s = edge < E ? ei[edge]     : edge - E;
  int d = edge < E ? ei[E + edge] : edge - E;
  float e = asn[s * 4 + h] + adn[d * 4 + h];
  e = e > 0.f ? e : 0.2f * e;
  atomicAdd(&sseg[d * 4 + h], expf(e - mseg[d * 4 + h]));
}

__global__ void gat_msg_kernel(const int* __restrict__ ei, int E, int N,
                               const float* __restrict__ tmp,
                               const float* __restrict__ asn,
                               const float* __restrict__ adn,
                               const float* __restrict__ mseg,
                               const float* __restrict__ sseg,
                               float* __restrict__ agg) {
  long long idx = (long long)blockIdx.x * blockDim.x + threadIdx.x;
  int edge = (int)(idx >> 5);
  int lane = (int)(idx & 31);
  if (edge >= E + N) return;
  int s = edge < E ? ei[edge]     : edge - E;
  int d = edge < E ? ei[E + edge] : edge - E;
  int h = lane >> 3;                          // lane's head
  float e = asn[s * 4 + h] + adn[d * 4 + h];
  e = e > 0.f ? e : 0.2f * e;
  float coef = expf(e - mseg[d * 4 + h]) / sseg[d * 4 + h];
  float4 v = ((const float4*)(tmp + (size_t)s * 128))[lane];
  float* a = agg + (size_t)d * 128 + lane * 4;
  atomicAdd(a + 0, v.x * coef);
  atomicAdd(a + 1, v.y * coef);
  atomicAdd(a + 2, v.z * coef);
  atomicAdd(a + 3, v.w * coef);
}

// ---------------------------------------------------------------------------
// BatchNorm: stats (sum / sumsq per channel) -> finalize -> apply+activation
// stats layout: [0:128) sum, [128:256) sumsq, [256:384) scale, [384:512) shift
// ---------------------------------------------------------------------------
__global__ void bn_stats_kernel(const float* __restrict__ h, int N,
                                float* __restrict__ stats) {
  int c = threadIdx.x & 127;
  int sub = threadIdx.x >> 7;                 // 2 rows per block pass
  float s = 0.f, q = 0.f;
  for (int r = blockIdx.x * 2 + sub; r < N; r += gridDim.x * 2) {
    float v = h[(size_t)r * 128 + c];
    s += v; q += v * v;
  }
  atomicAdd(&stats[c], s);
  atomicAdd(&stats[128 + c], q);
}

__global__ void bn_finalize_kernel(float* __restrict__ stats,
                                   const float* __restrict__ gamma,
                                   const float* __restrict__ beta, float invN) {
  int c = threadIdx.x;
  float mu  = stats[c] * invN;
  float var = stats[128 + c] * invN - mu * mu;  // biased variance
  float sc  = gamma[c] * rsqrtf(var + EPS);
  stats[256 + c] = sc;
  stats[384 + c] = beta[c] - mu * sc;
}

// act: 0 = relu, 1 = elu(alpha=1)
__global__ void bn_apply_kernel(const float* __restrict__ src,
                                const float* __restrict__ stats,
                                float* __restrict__ dst, int dstStride,
                                long long total, int act) {
  long long i = (long long)blockIdx.x * blockDim.x + threadIdx.x;
  if (i >= total) return;
  int c = (int)(i & 127);
  long long n = i >> 7;
  float v = src[i] * stats[256 + c] + stats[384 + c];
  v = (act == 0) ? (v > 0.f ? v : 0.f)
                 : (v > 0.f ? v : expf(v) - 1.f);
  dst[n * dstStride + c] = v;
}

__global__ void classifier_kernel(const float* __restrict__ h,
                                  const float* __restrict__ Wc,
                                  const float* __restrict__ bc,
                                  float* __restrict__ out, int N) {
  int i = blockIdx.x * blockDim.x + threadIdx.x;
  if (i >= N * 10) return;
  int n = i / 10, c = i % 10;
  const float* row = h + (size_t)n * 128;
  float acc = bc[c];
  for (int k = 0; k < 128; ++k) acc += row[k] * Wc[k * 10 + c];
  out[i] = acc;
}

// ---------------------------------------------------------------------------
static inline unsigned cdiv(long long a, int b) { return (unsigned)((a + b - 1) / b); }

static void run_bn(const float* src, float* stats, const float* gamma,
                   const float* beta, float* dst, int dstStride, int act,
                   int N, hipStream_t stream) {
  fill_kernel<<<1, 256, 0, stream>>>(stats, 0.f, 256);
  bn_stats_kernel<<<512, 256, 0, stream>>>(src, N, stats);
  bn_finalize_kernel<<<1, 128, 0, stream>>>(stats, gamma, beta, 1.f / (float)N);
  long long total = (long long)N * 128;
  bn_apply_kernel<<<cdiv(total, 256), 256, 0, stream>>>(src, stats, dst,
                                                        dstStride, total, act);
}

extern "C" void kernel_launch(void* const* d_in, const int* in_sizes, int n_in,
                              void* d_out, int out_size, void* d_ws, size_t ws_size,
                              hipStream_t stream) {
  const float* x         = (const float*)d_in[0];
  const int*   ei        = (const int*)  d_in[1];
  const float* gcn_W     = (const float*)d_in[2];
  const float* gcn_gamma = (const float*)d_in[4];
  const float* gcn_beta  = (const float*)d_in[5];
  const float* gat_W     = (const float*)d_in[6];
  const float* a_src     = (const float*)d_in[7];
  const float* a_dst     = (const float*)d_in[8];
  const float* gat_gamma = (const float*)d_in[10];
  const float* gat_beta  = (const float*)d_in[11];
  const float* fus_W     = (const float*)d_in[12];
  const float* fus_gamma = (const float*)d_in[14];
  const float* fus_beta  = (const float*)d_in[15];
  const float* cls_W     = (const float*)d_in[16];
  const float* cls_b     = (const float*)d_in[17];
  float* out = (float*)d_out;

  const int N = in_sizes[0] / HID;
  const int E = in_sizes[1] / 2;
  const int Etot = E + N;

  float* ws = (float*)d_ws;
  size_t off = 0;
  float* tmp    = ws + off; off += (size_t)N * HID;      // GEMM output
  float* agg    = ws + off; off += (size_t)N * HID;      // scatter target
  float* hbuf   = ws + off; off += (size_t)N * HID;      // inter-layer hidden
  float* concat = ws + off; off += (size_t)N * 2 * HID;  // [h_gcn | h_gat]
  float* deg    = ws + off; off += (size_t)N;
  float* dinv   = ws + off; off += (size_t)N;
  float* asn    = ws + off; off += (size_t)N * 4;
  float* adn    = ws + off; off += (size_t)N * 4;
  float* mseg   = ws + off; off += (size_t)N * 4;
  float* sseg   = ws + off; off += (size_t)N * 4;
  float* stats  = ws + off; off += 512;
  (void)ws_size; (void)n_in; (void)out_size;

  const long long NH   = (long long)N * HID;
  const unsigned gemmG = cdiv(N, 16);

  // --- shared degree normalization (self-loops included) ---
  fill_kernel<<<cdiv(N, 256), 256, 0, stream>>>(deg, 0.f, N);
  deg_kernel<<<cdiv(Etot, 256), 256, 0, stream>>>(ei, E, N, deg);
  dinv_kernel<<<cdiv(N, 256), 256, 0, stream>>>(deg, dinv, N);

  // --- GCN branch ---
  const float* hin = x;
  for (int i = 0; i < 2; ++i) {
    gemm_wmma_kernel<<<gemmG, 256, 0, stream>>>(hin, 128, 128,
                                                gcn_W + (size_t)i * 128 * 128, tmp, N);
    fill_kernel<<<cdiv(NH, 256), 256, 0, stream>>>(agg, 0.f, NH);
    gcn_scatter_kernel<<<cdiv((long long)Etot * 32, 256), 256, 0, stream>>>(
        ei, E, N, tmp, dinv, agg);
    float* dst = (i == 0) ? hbuf : concat;               // last layer -> concat[:, :128]
    run_bn(agg, stats, gcn_gamma + i * 128, gcn_beta + i * 128,
           dst, (i == 0) ? 128 : 256, /*relu*/0, N, stream);
    hin = hbuf;
  }

  // --- GAT branch ---
  hin = x;
  for (int i = 0; i < 2; ++i) {
    gemm_wmma_kernel<<<gemmG, 256, 0, stream>>>(hin, 128, 128,
                                                gat_W + (size_t)i * 128 * 128, tmp, N);
    gat_att_kernel<<<cdiv((long long)N * 4, 256), 256, 0, stream>>>(
        tmp, a_src + i * 128, a_dst + i * 128, asn, adn, N);
    fill_kernel<<<cdiv((long long)N * 4, 256), 256, 0, stream>>>(mseg, -FLT_MAX, (long long)N * 4);
    fill_kernel<<<cdiv((long long)N * 4, 256), 256, 0, stream>>>(sseg, 0.f, (long long)N * 4);
    gat_max_kernel<<<cdiv((long long)Etot * 4, 256), 256, 0, stream>>>(ei, E, N, asn, adn, mseg);
    gat_sum_kernel<<<cdiv((long long)Etot * 4, 256), 256, 0, stream>>>(ei, E, N, asn, adn, mseg, sseg);
    fill_kernel<<<cdiv(NH, 256), 256, 0, stream>>>(agg, 0.f, NH);
    gat_msg_kernel<<<cdiv((long long)Etot * 32, 256), 256, 0, stream>>>(
        ei, E, N, tmp, asn, adn, mseg, sseg, agg);
    float* dst = (i == 0) ? hbuf : (concat + 128);       // last layer -> concat[:, 128:]
    run_bn(agg, stats, gat_gamma + i * 128, gat_beta + i * 128,
           dst, (i == 0) ? 128 : 256, /*elu*/1, N, stream);
    hin = hbuf;
  }

  // --- fusion: [N,256] @ [256,128] -> BN -> relu ---
  gemm_wmma_kernel<<<gemmG, 256, 0, stream>>>(concat, 256, 256, fus_W, tmp, N);
  run_bn(tmp, stats, fus_gamma, fus_beta, hbuf, 128, /*relu*/0, N, stream);

  // --- classifier: [N,128] @ [128,10] + b ---
  classifier_kernel<<<cdiv((long long)N * 10, 256), 256, 0, stream>>>(hbuf, cls_W, cls_b, out, N);
}
